// RNNModel_73753178407315
// MI455X (gfx1250) — compile-verified, matrix operand order checked by reading, not run on hardware
//
#include <hip/hip_runtime.h>
#include <hip/hip_bf16.h>

// CDNA5 / gfx1250: wave32, WMMA 16x16x32 bf16 (f32 accumulate).

typedef __attribute__((ext_vector_type(16))) __bf16 v16bf;
typedef __attribute__((ext_vector_type(8)))  float  v8f;

#define HID   256     // HIDDEN
#define EMBED 100
#define KPAD  128     // EMBED padded to WMMA K granularity (4 k-tiles of 32)
#define TRN   128     // TRANSFORM
#define NCLS  4

union FragU {
    v16bf  v;
    __bf16 e[16];
    uint4  q[2];
};

// Build a B-matrix (K x N) bf16 fragment from a row-major f32 weight matrix.
// Lane L owns column n = ncol; element i covers k = kbase + khalf + (i&7) + (i>>3)*16,
// matching the ISA 16-bit operand layout (lanes 16-31 get khalf=8).
__device__ __forceinline__ v16bf make_bfrag_global(const float* __restrict__ W, int ldn,
                                                   int ncol, int kbase, int khalf, int kmax) {
    FragU f;
#pragma unroll
    for (int i = 0; i < 16; ++i) {
        int k = kbase + khalf + (i & 7) + ((i >> 3) << 4);
        float v = (k < kmax) ? W[k * ldn + ncol] : 0.0f;
        f.e[i] = (__bf16)v;
    }
    return f.v;
}

// Build an A-matrix fragment from LDS (row-major bf16, 8 elems = 1 uint4 per group).
// g0 selects the k-group for this lane's khalf; second half of the frag is g0+2.
__device__ __forceinline__ v16bf load_afrag(const uint4* __restrict__ row, int g0) {
    FragU f;
    f.q[0] = row[g0];
    f.q[1] = row[g0 + 2];
    return f.v;
}

// One workgroup = one 16-row batch tile of one RNN (blocks 0..15 body, 16..31 head).
// 16 waves; wave w owns output N-tile w (16 hidden columns). Weights stay in VGPRs
// as bf16 B-fragments; h ping-pongs in LDS as bf16. Two-deep software pipeline:
// tokens are staged one iteration ahead (tokA) so the embedding gather for t+1 has
// no in-iteration address dependency and drains only at the post-WMMA commit; one
// workgroup barrier per timestep.
__global__ __launch_bounds__(512)
void rnn_scan_kernel(const int* __restrict__ headlines,
                     const int* __restrict__ bodies,
                     const float* __restrict__ emb,
                     const float* __restrict__ Wx_head, const float* __restrict__ Wh_head,
                     const float* __restrict__ b_head,
                     const float* __restrict__ Wx_body, const float* __restrict__ Wh_body,
                     const float* __restrict__ b_body,
                     float* __restrict__ hout_head, float* __restrict__ hout_body) {
    __shared__ uint4 s_x[2][16 * (KPAD / 8)];  // ping-pong 16 x 128 bf16 x-tile (8 KB)
    __shared__ uint4 s_h[2][16 * (HID / 8)];   // ping-pong 16 x 256 bf16 hidden  (16 KB)

    const int  blk     = blockIdx.x;
    const bool is_head = (blk >= 16);
    const int  m0      = (blk & 15) * 16;          // batch rows m0..m0+15

    const int    T    = is_head ? 64 : 1024;
    const int*   toks = is_head ? headlines : bodies;
    const float* Wx   = is_head ? Wx_head : Wx_body;
    const float* Wh   = is_head ? Wh_head : Wh_body;
    const float* bv   = is_head ? b_head  : b_body;
    float*       hout = is_head ? hout_head : hout_body;

    const int tid   = threadIdx.x;
    const int lane  = tid & 31;
    const int wv    = tid >> 5;              // 0..15 : N-tile owned by this wave
    const int mrow  = lane & 15;             // A row / D column index within tile
    const int khalf = (lane >> 4) << 3;      // 0 or 8
    const int ghalf = khalf >> 3;            // 0 or 1
    const int ncol  = wv * 16 + mrow;        // hidden column owned for B / D

    // ---- preload weight B-fragments into VGPRs (bf16), once ----
    v16bf bwx[4];
#pragma unroll
    for (int kt = 0; kt < 4; ++kt)
        bwx[kt] = make_bfrag_global(Wx, HID, ncol, kt * 32, khalf, EMBED); // zero-pad K>=100
    v16bf bwh[8];
#pragma unroll
    for (int kt = 0; kt < 8; ++kt)
        bwh[kt] = make_bfrag_global(Wh, HID, ncol, kt * 32, khalf, HID);

    const float bias = bv[ncol];

    // gather work split: 32 threads per row, 4 consecutive elems per thread
    const int grow = tid >> 5;
    const int ge0  = (tid & 31) * 4;
    __bf16* xt = reinterpret_cast<__bf16*>(s_x);

    // ---- h(0) = 0 ----
    {
        __bf16* h0 = reinterpret_cast<__bf16*>(s_h[0]);
        for (int i = tid; i < 16 * HID; i += 512) h0[i] = (__bf16)0.0f;
    }
    // ---- stage x(0) into buffer 0, and token for t=1 into a register ----
    {
        const int tok = toks[(m0 + grow) * T + 0];
        float4 v = {0.0f, 0.0f, 0.0f, 0.0f};
        if (ge0 < EMBED)  // emb rows are 400B (16B-aligned), EMBED%4==0 -> safe float4
            v = *reinterpret_cast<const float4*>(emb + (size_t)tok * EMBED + ge0);
        __bf16* x0 = xt;
        x0[grow * KPAD + ge0 + 0] = (__bf16)v.x;
        x0[grow * KPAD + ge0 + 1] = (__bf16)v.y;
        x0[grow * KPAD + ge0 + 2] = (__bf16)v.z;
        x0[grow * KPAD + ge0 + 3] = (__bf16)v.w;
    }
    int tokA = (T > 1) ? toks[(m0 + grow) * T + 1] : 0;   // token for t+1
    __syncthreads();

    int cur = 0, xcur = 0;
    for (int t = 0; t < T; ++t) {
        // ---- issue gather of x(t+1) (address already resolved via tokA) and the
        //      t+2 token load; both drain at the commit AFTER the WMMA chain ----
        const bool have_next = (t + 1 < T);
        const bool have_nn   = (t + 2 < T);
        float4 nv = {0.0f, 0.0f, 0.0f, 0.0f};
        if (have_next && ge0 < EMBED)
            nv = *reinterpret_cast<const float4*>(emb + (size_t)tokA * EMBED + ge0);
        int tokB = 0;
        if (have_nn)
            tokB = toks[(m0 + grow) * T + t + 2];

        // ---- acc = bias + x(t) @ Wx + h(t) @ Wh   (12 WMMAs, f32 accumulate) ----
        v8f acc;
#pragma unroll
        for (int r = 0; r < 8; ++r) acc[r] = bias;

        const uint4* xrow = s_x[xcur] + mrow * (KPAD / 8);
#pragma unroll
        for (int kt = 0; kt < 4; ++kt) {
            v16bf a = load_afrag(xrow, kt * 4 + ghalf);
            acc = __builtin_amdgcn_wmma_f32_16x16x32_bf16(
                false, a, false, bwx[kt], (short)0, acc, false, false);
        }
        const uint4* hrow = s_h[cur] + mrow * (HID / 8);
#pragma unroll
        for (int kt = 0; kt < 8; ++kt) {
            v16bf a = load_afrag(hrow, kt * 4 + ghalf);
            acc = __builtin_amdgcn_wmma_f32_16x16x32_bf16(
                false, a, false, bwh[kt], (short)0, acc, false, false);
        }

        // ---- sigmoid + write h(t+1) tile (D layout: m = r + (lane>>4)*8, n = ncol) ----
        __bf16* hnext = reinterpret_cast<__bf16*>(s_h[cur ^ 1]);
        const bool last = (t == T - 1);
#pragma unroll
        for (int r = 0; r < 8; ++r) {
            const int m = r + ((lane >> 4) << 3);
            const float hv = 1.0f / (1.0f + __expf(-acc[r]));
            hnext[m * HID + ncol] = (__bf16)hv;
            if (last) hout[(size_t)(m0 + m) * HID + ncol] = hv;
        }

        // ---- commit staged x(t+1); loadcnt drains here, hidden by the WMMAs ----
        if (have_next) {
            __bf16* xn = xt + (xcur ^ 1) * (16 * KPAD);
            xn[grow * KPAD + ge0 + 0] = (__bf16)nv.x;
            xn[grow * KPAD + ge0 + 1] = (__bf16)nv.y;
            xn[grow * KPAD + ge0 + 2] = (__bf16)nv.z;
            xn[grow * KPAD + ge0 + 3] = (__bf16)nv.w;
        }
        // tokB is resolved by now (same loadcnt drain): prefetch t+2's row stall-free
        if (have_nn && ge0 == 0)
            __builtin_prefetch(emb + (size_t)tokB * EMBED, 0, 0);

        // single barrier per step: x(t)/h(t) reads, h(t+1)/x(t+1) writes use
        // disjoint buffers within an iteration.
        __syncthreads();
        tokA = tokB;
        cur ^= 1;
        xcur ^= 1;
    }
}

// V[side][k][c] = sum_j U_side[k][j] * W_pred[side*128 + j][c]   (2 x 256 x 4)
__global__ void fold_pred_weights(const float* __restrict__ U_head,
                                  const float* __restrict__ U_body,
                                  const float* __restrict__ W_pred,
                                  float* __restrict__ V) {
    const int id = blockIdx.x * blockDim.x + threadIdx.x;   // 0..2047
    if (id >= 2 * HID * NCLS) return;
    const int side = id >> 10;
    const int rem  = id & 1023;
    const int k    = rem >> 2;
    const int c    = rem & 3;
    const float* U = side ? U_body : U_head;
    float s = 0.0f;
    for (int j = 0; j < TRN; ++j)
        s += U[k * TRN + j] * W_pred[(side * TRN + j) * NCLS + c];
    V[id] = s;
}

// preds[b][c] = h_head[b] . V[0][:,c] + h_body[b] . V[1][:,c]
__global__ void pred_kernel(const float* __restrict__ h_head,
                            const float* __restrict__ h_body,
                            const float* __restrict__ V,
                            float* __restrict__ out) {
    const int id = blockIdx.x * blockDim.x + threadIdx.x;   // 0..1023
    const int b = id >> 2, c = id & 3;
    float s = 0.0f;
    for (int k = 0; k < HID; ++k) {
        s += h_head[b * HID + k] * V[(0 * HID + k) * NCLS + c];
        s += h_body[b * HID + k] * V[(1 * HID + k) * NCLS + c];
    }
    out[id] = s;
}

extern "C" void kernel_launch(void* const* d_in, const int* in_sizes, int n_in,
                              void* d_out, int out_size, void* d_ws, size_t ws_size,
                              hipStream_t stream) {
    const int*   headlines = (const int*)  d_in[0];
    const int*   bodies    = (const int*)  d_in[1];
    const float* emb       = (const float*)d_in[2];
    const float* Wx_head   = (const float*)d_in[3];
    const float* Wh_head   = (const float*)d_in[4];
    const float* b_head    = (const float*)d_in[5];
    const float* Wx_body   = (const float*)d_in[6];
    const float* Wh_body   = (const float*)d_in[7];
    const float* b_body    = (const float*)d_in[8];
    const float* U_head    = (const float*)d_in[9];
    const float* U_body    = (const float*)d_in[10];
    const float* W_pred    = (const float*)d_in[11];

    float* ws       = (float*)d_ws;
    float* h_head_f = ws;                    // 256*256 f32
    float* h_body_f = ws + 256 * 256;        // 256*256 f32
    float* V        = ws + 2 * 256 * 256;    // 2*256*4 f32

    // 32 independent persistent workgroups: 16 body tiles + 16 head tiles.
    rnn_scan_kernel<<<dim3(32), dim3(512), 0, stream>>>(
        headlines, bodies, emb,
        Wx_head, Wh_head, b_head,
        Wx_body, Wh_body, b_body,
        h_head_f, h_body_f);

    fold_pred_weights<<<dim3(8), dim3(256), 0, stream>>>(U_head, U_body, W_pred, V);
    pred_kernel<<<dim3(4), dim3(256), 0, stream>>>(h_head_f, h_body_f, V, (float*)d_out);
}
